// MyLSTM_5274219840209
// MI455X (gfx1250) — matrix operand order repaired and checked by reading, not compiled, not run on hardware
//
#include <hip/hip_runtime.h>

// LSTM: S=512, B=64, NIN=256, NHID=512.
// Persistent-kernel design: one kernel runs all 512 timesteps, separated by a
// device-wide atomic barrier (avoids 512 graph-node launch overheads on a
// latency-bound recurrence). Per step, gates = [x_t ; h] @ [W ; U] + b is a
// [64,768]x[768,2048] bf16 WMMA GEMM (v_wmma_f32_16x16x32_bf16), f32 accum.
// 512 waves <-> 512 (gate, m-tile, n-tile) 16x16 output tiles.

#define S_LEN 512
#define BATCH 64
#define NIN   256
#define NH    512
#define NG    2048          // 4*NH gate columns
#define KTOT  768           // NIN + NH
#define KIT   24            // KTOT / 32
#define NBLK  128           // 4 m-tiles * 32 h-tiles
#define BPAD  784           // KTOT + 16 bf16 pad (LDS bank-conflict mitigation)

typedef __attribute__((ext_vector_type(16))) __bf16 v16bf;
typedef __attribute__((ext_vector_type(8)))  __bf16 v8bf;
typedef __attribute__((ext_vector_type(8)))  float  v8f;

static __device__ __forceinline__ unsigned short f2bf(float f) {
  unsigned u = __float_as_uint(f);
  u += 0x7FFFu + ((u >> 16) & 1u);          // round-to-nearest-even
  return (unsigned short)(u >> 16);
}
static __device__ __forceinline__ float sigmoid_(float x) {
  return 1.0f / (1.0f + __expf(-x));
}
static __device__ __forceinline__ float tanh_(float x) {
  x = fminf(fmaxf(x, -15.0f), 15.0f);
  float e = __expf(2.0f * x);
  return (e - 1.0f) / (e + 1.0f);
}

// ---------------- prep kernels ----------------

__global__ void k_cvt_x(const float* __restrict__ x, unsigned short* __restrict__ xb, int n) {
  int t = blockIdx.x * blockDim.x + threadIdx.x;
  if (t < n) xb[t] = f2bf(x[t]);
}

// wT[c][k] (c in [0,2048), k in [0,768)) = bf16 of (k<256 ? W[k][c] : U[k-256][c])
__global__ void k_build_wT(const float* __restrict__ W, const float* __restrict__ U,
                           unsigned short* __restrict__ wT) {
  int t = blockIdx.x * blockDim.x + threadIdx.x;
  if (t >= NG * KTOT) return;
  int c = t / KTOT;
  int k = t % KTOT;
  float v = (k < NIN) ? W[(size_t)k * NG + c] : U[(size_t)(k - NIN) * NG + c];
  wT[t] = f2bf(v);
}

__global__ void k_init(unsigned short* __restrict__ h0, unsigned short* __restrict__ h1,
                       float* __restrict__ c, unsigned* __restrict__ bar) {
  int t = blockIdx.x * blockDim.x + threadIdx.x;
  if (t < BATCH * NH) { h0[t] = 0; h1[t] = 0; c[t] = 0.0f; }
  if (t < 8) bar[t] = 0u;
}

// ---------------- device-wide barrier (sense reversal) ----------------

static __device__ __forceinline__ void grid_barrier(unsigned* cnt, unsigned* gen) {
  __threadfence();            // publish h_bf16 writes device-wide
  __syncthreads();
  if (threadIdx.x == 0) {
    unsigned g = __hip_atomic_load(gen, __ATOMIC_ACQUIRE, __HIP_MEMORY_SCOPE_AGENT);
    unsigned a = __hip_atomic_fetch_add(cnt, 1u, __ATOMIC_ACQ_REL, __HIP_MEMORY_SCOPE_AGENT);
    if (a == (unsigned)(NBLK - 1)) {
      __hip_atomic_store(cnt, 0u, __ATOMIC_RELAXED, __HIP_MEMORY_SCOPE_AGENT);
      __hip_atomic_fetch_add(gen, 1u, __ATOMIC_RELEASE, __HIP_MEMORY_SCOPE_AGENT);
    } else {
      while (__hip_atomic_load(gen, __ATOMIC_ACQUIRE, __HIP_MEMORY_SCOPE_AGENT) == g)
        __builtin_amdgcn_s_sleep(2);
    }
  }
  __syncthreads();
}

// ---------------- persistent LSTM kernel ----------------

__global__ __launch_bounds__(128) void k_lstm(
    const unsigned short* __restrict__ xb,   // [S][B][NIN] bf16
    const unsigned short* __restrict__ wT,   // [2048][768] bf16 ([W;U] transposed)
    const float* __restrict__ bias,          // [2048]
    unsigned short* __restrict__ h0,         // [B][NH] bf16 double buffer
    unsigned short* __restrict__ h1,
    float* __restrict__ cst,                 // [B][NH] f32 cell state
    float* __restrict__ out,                 // [S][B][NH] ++ h_t[B][NH] ++ c_t[B][NH]
    unsigned* __restrict__ bar)
{
  __shared__ __align__(32) unsigned short bcache[4][16][BPAD]; // per-block weight cols
  __shared__ float glds[4][16][17];                            // gate tiles i/f/g/o

  const int tid  = threadIdx.x;
  const int wave = tid >> 5;            // gate index 0..3 (i,f,g,o)
  const int lane = tid & 31;
  const int mt   = blockIdx.x & 3;      // batch tile  (4  x 16 rows)
  const int ht   = blockIdx.x >> 2;     // hidden tile (32 x 16 cols)
  const int half = lane >> 4;
  const int l16  = lane & 15;

  // ---- stage this block's 64 weight columns (4 gates x 16 cols x 768 k) in LDS ----
  {
    const int nchunk = KTOT / 8;        // 96 x uint4 per row
    for (int t = tid; t < 64 * nchunk; t += 128) {
      int row = t / nchunk, ch = t % nchunk;
      int g = row >> 4, cc = row & 15;
      const uint4* src =
          (const uint4*)(wT + ((size_t)(g * NH + ht * 16 + cc)) * KTOT + ch * 8);
      *(uint4*)&bcache[g][cc][ch * 8] = *src;
    }
  }
  __syncthreads();

  const int r   = mt * 16 + l16;               // batch row this lane supplies (A)
  const int col = wave * NH + ht * 16 + l16;   // gate column this lane owns (B/C)
  const float bv = bias[col];

  float* out_h = out + (size_t)S_LEN * BATCH * NH;
  float* out_c = out_h + BATCH * NH;

  for (int s = 0; s < S_LEN; ++s) {
    const unsigned short* hread  = (s & 1) ? h1 : h0;
    unsigned short*       hwrite = (s & 1) ? h0 : h1;

    const unsigned short* xrow = xb + ((size_t)s * BATCH + r) * NIN;
    const unsigned short* hrow = hread + r * NH;

    v8f acc = {};
    #pragma unroll
    for (int it = 0; it < KIT; ++it) {
      const int kb = it * 32;
      // A fragment (per-lane contiguous K = kb+16*half .. +15, matching B)
      const unsigned short* arow = (it < 8) ? (xrow + kb) : (hrow + (kb - NIN));
      v16bf afrag = *(const v16bf*)(arow + 16 * half);
      // B fragment: lane owns column `col`, K = kb+16*half .. +15 (contiguous in LDS)
      v16bf bfrag = *(const v16bf*)&bcache[wave][l16][kb + 16 * half];
      acc = __builtin_amdgcn_wmma_f32_16x16x32_bf16(
          false, afrag, false, bfrag, (short)0, acc, false, false);
    }

    // C/D layout: VGPR i -> (M = i + 8*half, N = l16). Publish gate tile to LDS.
    #pragma unroll
    for (int i = 0; i < 8; ++i)
      glds[wave][i + 8 * half][l16] = acc[i] + bv;
    __syncthreads();

    // Elementwise LSTM cell: 256 tile elements / 128 threads = 2 each.
    #pragma unroll
    for (int e = 0; e < 2; ++e) {
      int idx = tid + e * 128;
      int m = idx >> 4, n = idx & 15;
      int gr = mt * 16 + m;          // batch index
      int gc = ht * 16 + n;          // hidden index
      float iv = sigmoid_(glds[0][m][n]);
      float fv = sigmoid_(glds[1][m][n]);
      float gv = tanh_   (glds[2][m][n]);
      float ov = sigmoid_(glds[3][m][n]);
      float cold = cst[gr * NH + gc];
      float cnew = fv * cold + iv * gv;
      float hnew = ov * tanh_(cnew);
      cst[gr * NH + gc] = cnew;
      out[((size_t)s * BATCH + gr) * NH + gc] = hnew;
      hwrite[gr * NH + gc] = f2bf(hnew);
      if (s == S_LEN - 1) {
        out_h[gr * NH + gc] = hnew;
        out_c[gr * NH + gc] = cnew;
      }
    }

    grid_barrier(bar, bar + 1);   // all h writes visible before next step reads
  }
}

// ---------------- host launch ----------------

extern "C" void kernel_launch(void* const* d_in, const int* in_sizes, int n_in,
                              void* d_out, int out_size, void* d_ws, size_t ws_size,
                              hipStream_t stream) {
  const float* x = (const float*)d_in[0];   // [S,B,NIN]
  const float* W = (const float*)d_in[1];   // [NIN,4H]
  const float* U = (const float*)d_in[2];   // [NH,4H]
  const float* b = (const float*)d_in[3];   // [4H]
  float* out = (float*)d_out;

  char* ws = (char*)d_ws;
  unsigned short* xb = (unsigned short*)(ws);                         // 16,777,216 B
  unsigned short* wT = (unsigned short*)(ws + 16777216);              //  3,145,728 B
  unsigned short* h0 = (unsigned short*)(ws + 19922944);              //     65,536 B
  unsigned short* h1 = (unsigned short*)(ws + 19988480);              //     65,536 B
  float*          c  = (float*)        (ws + 20054016);               //    131,072 B
  unsigned*       bar= (unsigned*)     (ws + 20185088);               //        256 B

  const int nx = S_LEN * BATCH * NIN;                 // 8,388,608
  k_cvt_x  <<<dim3((nx + 255) / 256),              dim3(256), 0, stream>>>(x, xb, nx);
  k_build_wT<<<dim3((NG * KTOT + 255) / 256),      dim3(256), 0, stream>>>(W, U, wT);
  k_init   <<<dim3((BATCH * NH + 255) / 256),      dim3(256), 0, stream>>>(h0, h1, c, bar);
  k_lstm   <<<dim3(NBLK),                          dim3(128), 0, stream>>>(
      xb, wT, b, h0, h1, c, out, bar);
}
